// ShiftedWindowAttention_22462678958665
// MI455X (gfx1250) — matrix operand order repaired and checked by reading, not compiled, not run on hardware
//
#include <hip/hip_runtime.h>
#include <hip/hip_bf16.h>

typedef __attribute__((ext_vector_type(16))) _Float16 v16h;
typedef __attribute__((ext_vector_type(8)))  float    v8f;

#define NHEAD 4
#define HD    32
#define WSZ   7
#define WS2   49
#define SSZ   3
#define IMG   56
#define BATCH 32
#define NWIN  64                       // windows per batch image (8x8)
#define TOTAL_WIN  (BATCH * NWIN)      // 2048
#define TOTAL_ROWS (TOTAL_WIN * WS2)   // 100352
#define CDIM  128
#define C3    384

// ---------------- WMMA fragment builders (CDNA5 wave32 layouts) ----------------
// A (16x32 f16): lanes 0-15 row M=lane, hold K=0..15, lanes 16-31 (row M=lane-16) hold K=16..31,
// element s is K=16*(lane/16)+s, matching the B builders.
__device__ inline v16h frag_a_lds(const _Float16* A, int ld) {
  int lane = threadIdx.x & 31;
  int hl = lane >> 4, l = lane & 15;
  v16h a;
#pragma unroll
  for (int v = 0; v < 8; ++v) {
    int kb = (v << 1) + (hl << 4);
    a[2 * v]     = A[l * ld + kb];
    a[2 * v + 1] = A[l * ld + kb + 1];
  }
  return a;
}

// B (32x16 f16) from row-major B[k][n] at B[k*ld+n]: lanes 0-15 hold K=0..15, lanes 16-31 K=16..31.
__device__ inline v16h frag_b_lds(const _Float16* B, int ld) {
  int lane = threadIdx.x & 31;
  int hl = lane >> 4, l = lane & 15;
  v16h b;
#pragma unroll
  for (int v = 0; v < 8; ++v) {
    int k = (v << 1) + (hl << 4);
    b[2 * v]     = B[k * ld + l];
    b[2 * v + 1] = B[(k + 1) * ld + l];
  }
  return b;
}

// B (32x16) where B[k][n] = Bt[n*ld + k]  (i.e. B = Bt^T, Bt row-major) — f16 source (LDS).
__device__ inline v16h frag_bT_lds(const _Float16* Bt, int ld) {
  int lane = threadIdx.x & 31;
  int hl = lane >> 4, l = lane & 15;
  const _Float16* row = Bt + l * ld + (hl << 4);
  v16h b;
#pragma unroll
  for (int v = 0; v < 8; ++v) {
    b[2 * v]     = row[2 * v];
    b[2 * v + 1] = row[2 * v + 1];
  }
  return b;
}

// B (32x16) where B[k][n] = W[(n0+n)*ld + k0+k], fp32 global weight, converted to f16.
// Each lane reads 16 contiguous floats (64B) — coalesced, L2-resident weights.
__device__ inline v16h frag_bT_gf32(const float* W, int n0, int k0, int ld) {
  int lane = threadIdx.x & 31;
  int hl = lane >> 4, l = lane & 15;
  const float* row = W + (size_t)(n0 + l) * ld + k0 + (hl << 4);
  v16h b;
#pragma unroll
  for (int v = 0; v < 8; ++v) {
    b[2 * v]     = (_Float16)row[2 * v];
    b[2 * v + 1] = (_Float16)row[2 * v + 1];
  }
  return b;
}

__device__ inline int swin_region(int p) {
  // canvas slices [0,49), [49,53), [53,56)  (ws=7, shift=3)
  return (p < IMG - WSZ) ? 0 : ((p < IMG - SSZ) ? 1 : 2);
}

// =============== Kernel 1: shifted-window gather + QKV GEMM (f16 out) ===============
// grid (1568, 6), block 128. Block computes rows [bx*64, +64) x cols [by*64, +64) of
// qkv = Xw @ qkv_weight^T + bias, q scaled by hd^-1/2.
__global__ __launch_bounds__(128) void swin_qkv_kernel(
    const float* __restrict__ x, const float* __restrict__ qkv_w,
    const float* __restrict__ qkv_b, _Float16* __restrict__ qkv16) {
  __shared__ _Float16 As[64][136];  // 64x128 f16, stride 136 to avoid bank conflicts
  int tid = threadIdx.x;
  {
    int lr = tid >> 1;
    int koff = (tid & 1) << 6;
    int r = blockIdx.x * 64 + lr;               // < 100352 exactly
    int wdx = r / WS2, t = r % WS2;
    int b = wdx >> 6, wh = (wdx >> 3) & 7, ww = wdx & 7;
    int hh = (wh * WSZ + t / WSZ + SSZ) % IMG;  // roll(-shift): canvas p reads x[(p+3)%56]
    int wc = (ww * WSZ + t % WSZ + SSZ) % IMG;
    const float4* s4 = (const float4*)(x + (((size_t)b * IMG + hh) * IMG + wc) * CDIM + koff);
#pragma unroll
    for (int i = 0; i < 16; ++i) {
      float4 f = s4[i];
      As[lr][koff + 4 * i + 0] = (_Float16)f.x;
      As[lr][koff + 4 * i + 1] = (_Float16)f.y;
      As[lr][koff + 4 * i + 2] = (_Float16)f.z;
      As[lr][koff + 4 * i + 3] = (_Float16)f.w;
    }
  }
  __syncthreads();

  int wave = tid >> 5, lane = tid & 31, hl = lane >> 4, l = lane & 15;
  v8f zero = {0.f, 0.f, 0.f, 0.f, 0.f, 0.f, 0.f, 0.f};
  v8f acc[4] = {zero, zero, zero, zero};
#pragma unroll
  for (int kk = 0; kk < 4; ++kk) {
    v16h a = frag_a_lds(&As[wave * 16][kk * 32], 136);
#pragma unroll
    for (int nt = 0; nt < 4; ++nt) {
      v16h b = frag_bT_gf32(qkv_w, blockIdx.y * 64 + nt * 16, kk * 32, CDIM);
      acc[nt] = __builtin_amdgcn_wmma_f32_16x16x32_f16(false, a, false, b, (short)0,
                                                       acc[nt], false, false);
    }
  }
#pragma unroll
  for (int nt = 0; nt < 4; ++nt) {
    int col = blockIdx.y * 64 + nt * 16 + l;
    float bias = qkv_b[col];
    float scale = (col < CDIM) ? 0.17677669529663689f : 1.0f;  // q * hd^-0.5
#pragma unroll
    for (int v = 0; v < 8; ++v) {
      size_t row = (size_t)blockIdx.x * 64 + wave * 16 + v + 8 * hl;
      qkv16[row * C3 + col] = (_Float16)((acc[nt][v] + bias) * scale);
    }
  }
}

// =============== Kernel 2: per-(window,head) attention ===============
// grid (2048, 4), block 128 (4 waves). 49 tokens padded to 64.
__global__ __launch_bounds__(128) void swin_attn_kernel(
    const _Float16* __restrict__ qkv16, const float* __restrict__ rpb_table,
    _Float16* __restrict__ aout) {
  __shared__ _Float16 qs[64][40], ksm[64][40], vsm[64][40];
  __shared__ _Float16 ps[64][72];
  int w = blockIdx.x, head = blockIdx.y;
  int tid = threadIdx.x;

  for (int idx = tid; idx < 64 * HD; idx += 128) {
    int t = idx >> 5, d = idx & 31;
    _Float16 qv = (_Float16)0.f, kv = qv, vv = qv;
    if (t < WS2) {
      size_t base = ((size_t)w * WS2 + t) * C3 + head * HD + d;
      qv = qkv16[base];
      kv = qkv16[base + CDIM];
      vv = qkv16[base + 2 * CDIM];
    }
    qs[t][d] = qv; ksm[t][d] = kv; vsm[t][d] = vv;
  }
  __syncthreads();

  int wave = tid >> 5, lane = tid & 31, hl = lane >> 4, l = lane & 15;
  int winh = (w >> 3) & 7, winw = w & 7;

  // scores = q @ k^T : one K=32 WMMA per 16x16 tile; wave handles M-tile = wave, 4 N-tiles
  v8f zero = {0.f, 0.f, 0.f, 0.f, 0.f, 0.f, 0.f, 0.f};
  v8f acc[4] = {zero, zero, zero, zero};
  {
    v16h a = frag_a_lds(&qs[wave * 16][0], 40);
#pragma unroll
    for (int nt = 0; nt < 4; ++nt) {
      v16h b = frag_bT_lds(&ksm[nt * 16][0], 40);  // B[k=d][n=t2] = K[t2][d]
      acc[nt] = __builtin_amdgcn_wmma_f32_16x16x32_f16(false, a, false, b, (short)0,
                                                       acc[nt], false, false);
    }
  }
  // + relative position bias, + shift mask, -inf for padded key columns
#pragma unroll
  for (int nt = 0; nt < 4; ++nt) {
#pragma unroll
    for (int v = 0; v < 8; ++v) {
      int t1 = wave * 16 + v + 8 * hl;
      int t2 = nt * 16 + l;
      float s = acc[nt][v];
      if (t2 >= WS2) {
        s = -1e30f;
      } else if (t1 < WS2) {
        int qh = t1 / WSZ, qw = t1 % WSZ, kh = t2 / WSZ, kw = t2 % WSZ;
        s += rpb_table[((qh - kh + WSZ - 1) * (2 * WSZ - 1) + (qw - kw + WSZ - 1)) * NHEAD + head];
        int lab1 = swin_region(winh * WSZ + qh) * 3 + swin_region(winw * WSZ + qw);
        int lab2 = swin_region(winh * WSZ + kh) * 3 + swin_region(winw * WSZ + kw);
        if (lab1 != lab2) s += -100.0f;
      }
      acc[nt][v] = s;
    }
  }
  // softmax: each row lives in one half-wave (16 lanes) x 4 accumulators
#pragma unroll
  for (int v = 0; v < 8; ++v) {
    float s0 = acc[0][v], s1 = acc[1][v], s2 = acc[2][v], s3 = acc[3][v];
    float m = fmaxf(fmaxf(s0, s1), fmaxf(s2, s3));
#pragma unroll
    for (int off = 1; off < 16; off <<= 1) m = fmaxf(m, __shfl_xor(m, off, 32));
    float e0 = __expf(s0 - m), e1 = __expf(s1 - m), e2 = __expf(s2 - m), e3 = __expf(s3 - m);
    float ssum = e0 + e1 + e2 + e3;
#pragma unroll
    for (int off = 1; off < 16; off <<= 1) ssum += __shfl_xor(ssum, off, 32);
    float inv = 1.0f / ssum;
    int t1 = wave * 16 + v + 8 * hl;
    ps[t1][ 0 + l] = (_Float16)(e0 * inv);
    ps[t1][16 + l] = (_Float16)(e1 * inv);
    ps[t1][32 + l] = (_Float16)(e2 * inv);
    ps[t1][48 + l] = (_Float16)(e3 * inv);
  }
  __syncthreads();

  // out = P @ V : K = 64 (2 steps), N = 32 (2 tiles)
  v8f acc2[2] = {zero, zero};
#pragma unroll
  for (int kk = 0; kk < 2; ++kk) {
    v16h pa = frag_a_lds(&ps[wave * 16][kk * 32], 72);
#pragma unroll
    for (int nt = 0; nt < 2; ++nt) {
      v16h b = frag_b_lds(&vsm[kk * 32][nt * 16], 40);  // B[k=t2][n=d] = V[t2][d]
      acc2[nt] = __builtin_amdgcn_wmma_f32_16x16x32_f16(false, pa, false, b, (short)0,
                                                        acc2[nt], false, false);
    }
  }
#pragma unroll
  for (int nt = 0; nt < 2; ++nt) {
#pragma unroll
    for (int v = 0; v < 8; ++v) {
      int t1 = wave * 16 + v + 8 * hl;
      if (t1 < WS2) {
        int d = nt * 16 + l;
        aout[((size_t)w * WS2 + t1) * CDIM + head * HD + d] = (_Float16)acc2[nt][v];
      }
    }
  }
}

// =============== Kernel 3: output projection + reverse-roll scatter ===============
// grid (1568, 2), block 128.
__global__ __launch_bounds__(128) void swin_proj_kernel(
    const _Float16* __restrict__ aout, const float* __restrict__ proj_w,
    const float* __restrict__ proj_b, float* __restrict__ out) {
  __shared__ _Float16 As[64][136];
  int tid = threadIdx.x;
  {
    int lr = tid >> 1;
    int koff = (tid & 1) << 6;
    const uint32_t* src = (const uint32_t*)(aout + ((size_t)blockIdx.x * 64 + lr) * CDIM + koff);
    uint32_t* dst = (uint32_t*)&As[lr][koff];
#pragma unroll
    for (int i = 0; i < 32; ++i) dst[i] = src[i];
  }
  __syncthreads();

  int wave = tid >> 5, lane = tid & 31, hl = lane >> 4, l = lane & 15;
  v8f zero = {0.f, 0.f, 0.f, 0.f, 0.f, 0.f, 0.f, 0.f};
  v8f acc[4] = {zero, zero, zero, zero};
#pragma unroll
  for (int kk = 0; kk < 4; ++kk) {
    v16h a = frag_a_lds(&As[wave * 16][kk * 32], 136);
#pragma unroll
    for (int nt = 0; nt < 4; ++nt) {
      v16h b = frag_bT_gf32(proj_w, blockIdx.y * 64 + nt * 16, kk * 32, CDIM);
      acc[nt] = __builtin_amdgcn_wmma_f32_16x16x32_f16(false, a, false, b, (short)0,
                                                       acc[nt], false, false);
    }
  }
#pragma unroll
  for (int nt = 0; nt < 4; ++nt) {
    int col = blockIdx.y * 64 + nt * 16 + l;
    float bias = proj_b[col];
#pragma unroll
    for (int v = 0; v < 8; ++v) {
      int r = blockIdx.x * 64 + wave * 16 + v + 8 * hl;
      int wdx = r / WS2, t = r % WS2;
      int b = wdx >> 6, wh = (wdx >> 3) & 7, ww = wdx & 7;
      int hh = (wh * WSZ + t / WSZ + SSZ) % IMG;  // canvas p -> final (p+3)%56 (roll(+shift))
      int wc = (ww * WSZ + t % WSZ + SSZ) % IMG;
      out[(((size_t)b * IMG + hh) * IMG + wc) * CDIM + col] = acc[nt][v] + bias;
    }
  }
}

extern "C" void kernel_launch(void* const* d_in, const int* in_sizes, int n_in,
                              void* d_out, int out_size, void* d_ws, size_t ws_size,
                              hipStream_t stream) {
  (void)in_sizes; (void)n_in; (void)out_size; (void)ws_size;
  const float* x      = (const float*)d_in[0];
  const float* qkv_w  = (const float*)d_in[1];
  const float* qkv_b  = (const float*)d_in[2];
  const float* proj_w = (const float*)d_in[3];
  const float* proj_b = (const float*)d_in[4];
  const float* rpb    = (const float*)d_in[5];
  float* out = (float*)d_out;

  _Float16* qkv16 = (_Float16*)d_ws;                           // [100352, 384] f16
  _Float16* aout  = qkv16 + (size_t)TOTAL_ROWS * C3;           // [100352, 128] f16

  dim3 g1(TOTAL_ROWS / 64, C3 / 64);   // (1568, 6)
  swin_qkv_kernel<<<g1, 128, 0, stream>>>(x, qkv_w, qkv_b, qkv16);

  dim3 g2(TOTAL_WIN, NHEAD);           // (2048, 4)
  swin_attn_kernel<<<g2, 128, 0, stream>>>(qkv16, rpb, aout);

  dim3 g3(TOTAL_ROWS / 64, CDIM / 64); // (1568, 2)
  swin_proj_kernel<<<g3, 128, 0, stream>>>(aout, proj_w, proj_b, out);
}
